// LstmLayer_23708219474404
// MI455X (gfx1250) — matrix operand order repaired and checked by reading, not compiled, not run on hardware
//
#include <hip/hip_runtime.h>

// CDNA5 / gfx1250: wave32, WMMA f32<-f16 16x16x32.
typedef __attribute__((ext_vector_type(16))) _Float16 v16h;
typedef __attribute__((ext_vector_type(8)))  float    v8f;

#define LSTM_B 512
#define LSTM_T 2048
#define LSTM_F 32

__device__ __forceinline__ float sigmoid_f(float x) {
    return 1.0f / (1.0f + __expf(-x));
}
__device__ __forceinline__ float tanh_f(float x) {
    // tanh(|x|) = (1-e)/(1+e), e = exp(-2|x|); overflow-safe, sign restored.
    float a = fabsf(x);
    float e = __expf(-2.0f * a);
    float t = (1.0f - e) / (1.0f + e);
    return copysignf(t, x);
}

// One wave (32 lanes) handles 16 batch rows for the full T=2048 scan.
// Orientation: G^T = W * [x|h]^T, so weights are WMMA A-operands (resident in
// VGPRs) and activations are B-operands. C/D layout (m=feature, n=batch) maps
// to the next step's B layout with one half-wave shfl_xor(16) per register.
__global__ __launch_bounds__(32)
void lstm_fused_wmma(const float* __restrict__ x,
                     const float* __restrict__ Wih,
                     const float* __restrict__ Whh,
                     const float* __restrict__ bih,
                     const float* __restrict__ bhh,
                     const float* __restrict__ Wlin,
                     const float* __restrict__ blin,
                     float* __restrict__ out)
{
    const int lane = threadIdx.x & 31;
    const int nlo  = lane & 15;   // batch within tile / tile column
    const int hi   = lane >> 4;   // half-wave flag
    const int koff = hi * 8;
    const int b0   = blockIdx.x * 16;

    // ---- A-format weight tiles (lane nlo = row m; k=16*hi+e) ----
    v16h WihA[8], WhhA[8];
    #pragma unroll
    for (int j = 0; j < 8; ++j) {
        const float* r0 = Wih + (j * 16 + nlo) * LSTM_F;
        const float* r1 = Whh + (j * 16 + nlo) * LSTM_F;
        #pragma unroll
        for (int e = 0; e < 8; ++e) {
            WihA[j][e]     = (_Float16)r0[2 * koff + e];
            WihA[j][e + 8] = (_Float16)r0[2 * koff + 8 + e];
            WhhA[j][e]     = (_Float16)r1[2 * koff + e];
            WhhA[j][e + 8] = (_Float16)r1[2 * koff + 8 + e];
        }
    }
    // Output linear, de-interleaved: even columns hit h, odd columns hit x.
    v16h WeA[2], WoA[2];
    #pragma unroll
    for (int jt = 0; jt < 2; ++jt) {
        const float* r = Wlin + (jt * 16 + nlo) * (2 * LSTM_F);
        #pragma unroll
        for (int e = 0; e < 16; ++e) {
            int k = e + 2 * koff;
            WeA[jt][e] = (_Float16)r[2 * k];
            WoA[jt][e] = (_Float16)r[2 * k + 1];
        }
    }
    // Biases in C/D layout: component v of tile j -> row 16j + v + 8*hi.
    v8f biasG[8];
    #pragma unroll
    for (int j = 0; j < 8; ++j)
        #pragma unroll
        for (int v = 0; v < 8; ++v) {
            int g = j * 16 + v + 8 * hi;
            biasG[j][v] = bih[g] + bhh[g];
        }
    v8f biasL[2];
    #pragma unroll
    for (int jt = 0; jt < 2; ++jt)
        #pragma unroll
        for (int v = 0; v < 8; ++v)
            biasL[jt][v] = blin[jt * 16 + v + 8 * hi];

    // ---- State ----
    v8f c0, c1;                       // cell, features 0-15 / 16-31 (m=feat, n=batch)
    #pragma unroll
    for (int v = 0; v < 8; ++v) { c0[v] = 0.0f; c1[v] = 0.0f; }
    v16h hB;                          // h^T in B layout (k=feature, n=batch)
    #pragma unroll
    for (int e = 0; e < 16; ++e) hB[e] = (_Float16)0.0f;

    // Per-lane streaming pointers: lane's batch row, feature half hi.
    const float* xrow = x   + (size_t)(b0 + nlo) * LSTM_T * LSTM_F + hi * 16;
    float*       orow = out + (size_t)(b0 + nlo) * LSTM_T * LSTM_F + hi * 8;

    // Prime x(t=0): 16 contiguous floats per lane.
    const float4* xp0 = (const float4*)xrow;
    float4 xq0 = xp0[0], xq1 = xp0[1], xq2 = xp0[2], xq3 = xp0[3];

    for (int t = 0; t < LSTM_T; ++t) {
        // x^T B-operand for this step.
        v16h xB;
        xB[0]  = (_Float16)xq0.x; xB[1]  = (_Float16)xq0.y;
        xB[2]  = (_Float16)xq0.z; xB[3]  = (_Float16)xq0.w;
        xB[4]  = (_Float16)xq1.x; xB[5]  = (_Float16)xq1.y;
        xB[6]  = (_Float16)xq1.z; xB[7]  = (_Float16)xq1.w;
        xB[8]  = (_Float16)xq2.x; xB[9]  = (_Float16)xq2.y;
        xB[10] = (_Float16)xq2.z; xB[11] = (_Float16)xq2.w;
        xB[12] = (_Float16)xq3.x; xB[13] = (_Float16)xq3.y;
        xB[14] = (_Float16)xq3.z; xB[15] = (_Float16)xq3.w;

        // Software-prefetch x(t+1); wait lands next iteration.
        int tn = (t + 1 < LSTM_T) ? (t + 1) : t;
        const float4* xpn = (const float4*)(xrow + (size_t)tn * LSTM_F);
        xq0 = xpn[0]; xq1 = xpn[1]; xq2 = xpn[2]; xq3 = xpn[3];

        // Gate tiles: j = half -> i, 2+half -> f, 4+half -> g, 6+half -> o.
        v8f hT0, hT1;
        #pragma unroll
        for (int half = 0; half < 2; ++half) {
            v8f ai = __builtin_amdgcn_wmma_f32_16x16x32_f16(false, WihA[0 + half], false, xB,
                                                            (short)0, biasG[0 + half], false, false);
            ai     = __builtin_amdgcn_wmma_f32_16x16x32_f16(false, WhhA[0 + half], false, hB,
                                                            (short)0, ai, false, false);
            v8f af = __builtin_amdgcn_wmma_f32_16x16x32_f16(false, WihA[2 + half], false, xB,
                                                            (short)0, biasG[2 + half], false, false);
            af     = __builtin_amdgcn_wmma_f32_16x16x32_f16(false, WhhA[2 + half], false, hB,
                                                            (short)0, af, false, false);
            v8f ag = __builtin_amdgcn_wmma_f32_16x16x32_f16(false, WihA[4 + half], false, xB,
                                                            (short)0, biasG[4 + half], false, false);
            ag     = __builtin_amdgcn_wmma_f32_16x16x32_f16(false, WhhA[4 + half], false, hB,
                                                            (short)0, ag, false, false);
            v8f ao = __builtin_amdgcn_wmma_f32_16x16x32_f16(false, WihA[6 + half], false, xB,
                                                            (short)0, biasG[6 + half], false, false);
            ao     = __builtin_amdgcn_wmma_f32_16x16x32_f16(false, WhhA[6 + half], false, hB,
                                                            (short)0, ao, false, false);

            v8f& cc = half ? c1 : c0;
            v8f& hh = half ? hT1 : hT0;
            #pragma unroll
            for (int v = 0; v < 8; ++v) {
                float iv = sigmoid_f(ai[v]);
                float fv = sigmoid_f(af[v]);
                float gv = tanh_f(ag[v]);
                float ov = sigmoid_f(ao[v]);
                float cv = fv * cc[v] + iv * gv;
                cc[v] = cv;
                hh[v] = ov * tanh_f(cv);
            }
        }

        // C/D layout -> B layout for h: one half-wave swap per register.
        #pragma unroll
        for (int v = 0; v < 8; ++v) {
            float a = hT0[v], b = hT1[v];
            float u = hi ? a : b;
            float w = __shfl_xor(u, 16, 32);
            hB[v]     = (_Float16)(hi ? w : a);   // k = v     (+16*hi)
            hB[v + 8] = (_Float16)(hi ? b : w);   // k = v + 8 (+16*hi)
        }

        // Fused output linear: out = tanh(h*We^T + x*Wo^T + b_lin).
        #pragma unroll
        for (int jt = 0; jt < 2; ++jt) {
            v8f oa = __builtin_amdgcn_wmma_f32_16x16x32_f16(false, WeA[jt], false, hB,
                                                            (short)0, biasL[jt], false, false);
            oa     = __builtin_amdgcn_wmma_f32_16x16x32_f16(false, WoA[jt], false, xB,
                                                            (short)0, oa, false, false);
            float4 s0, s1;
            s0.x = tanh_f(oa[0]); s0.y = tanh_f(oa[1]);
            s0.z = tanh_f(oa[2]); s0.w = tanh_f(oa[3]);
            s1.x = tanh_f(oa[4]); s1.y = tanh_f(oa[5]);
            s1.z = tanh_f(oa[6]); s1.w = tanh_f(oa[7]);
            float4* op = (float4*)(orow + (size_t)t * LSTM_F + jt * 16);
            op[0] = s0;
            op[1] = s1;
        }
    }
}

extern "C" void kernel_launch(void* const* d_in, const int* in_sizes, int n_in,
                              void* d_out, int out_size, void* d_ws, size_t ws_size,
                              hipStream_t stream) {
    const float* x    = (const float*)d_in[0];
    const float* Wih  = (const float*)d_in[1];
    const float* Whh  = (const float*)d_in[2];
    const float* bih  = (const float*)d_in[3];
    const float* bhh  = (const float*)d_in[4];
    const float* Wlin = (const float*)d_in[5];
    const float* blin = (const float*)d_in[6];
    float* out = (float*)d_out;

    dim3 grid(LSTM_B / 16);   // 32 waves, 16 batch rows each
    dim3 block(32);           // one wave32 per workgroup
    hipLaunchKernelGGL(lstm_fused_wmma, grid, block, 0, stream,
                       x, Wih, Whh, bih, bhh, Wlin, blin, out);
}
